// WorldModel_884763263653
// MI455X (gfx1250) — matrix operand order repaired and checked by reading, not compiled, not run on hardware
//
#include <hip/hip_runtime.h>

#define B_  2
#define N_  2048
#define D_  1024
#define H_  16
#define DH_ 64
#define BS_ 256

typedef _Float16 f16;
typedef __attribute__((ext_vector_type(16))) _Float16 v16h;
typedef __attribute__((ext_vector_type(8)))  float    v8f;

union V16 { v16h v; f16 h[16]; unsigned int u[8]; };

// ---------------------------------------------------------------------------
// Fragment loaders (wave32).
// A-matrix 16x32 f16 operand (this code's contiguous k order): lane L -> row M=L%16;
//   element e -> K = e + 16*(L>=16)                        => two 16B reads.
// B-matrix 32x16 f16 operand (same contiguous k order as A): lane L -> col N=L%16;
//   element e -> K = e + 16*(L>=16)                        => one 32B read.
// Both loaders take a row-major [row][k] tile (row = M for A, row = N for B
// when the tile is stored K-contiguous / transposed).
// ---------------------------------------------------------------------------
__device__ __forceinline__ v16h load_frag_a(const f16* p, int stride, int row,
                                            int kbase, int lane) {
  V16 u;
  const f16* r0 = p + (size_t)row * stride + kbase + ((lane >= 16) ? 16 : 0);
#pragma unroll
  for (int e = 0; e < 8; ++e) u.h[e] = r0[e];
  const f16* r1 = r0 + 8;
#pragma unroll
  for (int e = 0; e < 8; ++e) u.h[8 + e] = r1[e];
  return u.v;
}

__device__ __forceinline__ v16h load_frag_b(const f16* p, int stride, int row,
                                            int kbase, int lane) {
  V16 u;
  const f16* r0 = p + (size_t)row * stride + kbase + ((lane >= 16) ? 16 : 0);
#pragma unroll
  for (int e = 0; e < 16; ++e) u.h[e] = r0[e];
  return u.v;
}

__device__ __forceinline__ v8f wmma_f16(v16h a, v16h b, v8f c) {
  return __builtin_amdgcn_wmma_f32_16x16x32_f16(false, a, false, b,
                                                (short)0, c, false, false);
}

// ---------------------------------------------------------------------------
// RMSNorm: one block per (b,n) row, write f16 xn.
// ---------------------------------------------------------------------------
__global__ void k_rmsnorm(const float* __restrict__ x,
                          const float* __restrict__ gamma,
                          f16* __restrict__ xnh) {
  const int row = blockIdx.x;          // b*N + n
  const int tid = threadIdx.x;         // 256 threads
  __shared__ float red[8];
  const float* xr = x + (size_t)row * D_;
  float ss = 0.f;
  for (int d = tid; d < D_; d += 256) { float v = xr[d]; ss += v * v; }
#pragma unroll
  for (int m = 16; m; m >>= 1) ss += __shfl_xor(ss, m, 32);
  if ((tid & 31) == 0) red[tid >> 5] = ss;
  __syncthreads();
  float s2 = 0.f;
#pragma unroll
  for (int i = 0; i < 8; ++i) s2 += red[i];
  const float rinv = rsqrtf(s2 * (1.0f / D_) + 1.1920928955078125e-07f);
  for (int d = tid; d < D_; d += 256)
    xnh[(size_t)row * D_ + d] = (f16)(xr[d] * rinv * gamma[d]);
}

__global__ void k_f32_to_f16(const float* __restrict__ in, f16* __restrict__ out,
                             int n) {
  int i = blockIdx.x * blockDim.x + threadIdx.x;
  if (i < n) out[i] = (f16)in[i];
}

// ---------------------------------------------------------------------------
// Tiled f16 WMMA GEMM: C[M,Nn] = A[M,K] * B[K,Nn]. 64x64 tile, 128 threads
// (4 waves, one 16-row strip each). K staged in chunks of 64 (8 WMMA / stage).
// launch_bounds(128,1): max VGPR budget, no spills.
// ---------------------------------------------------------------------------
template <bool OUT_F16>
__global__ __launch_bounds__(128, 1)
void k_gemm(const f16* __restrict__ A, const f16* __restrict__ Bm,
            void* __restrict__ Cout, int M, int K, int Nn) {
  __shared__ f16 As[64 * 64];   // [m][k]
  __shared__ f16 Bt[64 * 64];   // [n][k] (transposed stage)
  const int tid  = threadIdx.x;
  const int lane = tid & 31;
  const int wave = tid >> 5;
  const int l16  = lane & 15;
  const int m0 = blockIdx.y * 64;
  const int n0 = blockIdx.x * 64;
  v8f acc[4] = {};
  for (int k0 = 0; k0 < K; k0 += 64) {
    __syncthreads();
    // Stage A tile: 64 rows x 64 k  (2048 dwords, coalesced)
    const unsigned int* Ag = (const unsigned int*)(A + (size_t)m0 * K + k0);
#pragma unroll
    for (int i = 0; i < 16; ++i) {
      int idx = tid + i * 128;               // dword index
      int r = idx >> 5, c = idx & 31;
      ((unsigned int*)As)[r * 32 + c] = Ag[(size_t)r * (K >> 1) + c];
    }
    // Stage B tile transposed: read [k][n] coalesced, write [n][k]
    const f16* Bg = Bm + (size_t)k0 * Nn + n0;
#pragma unroll
    for (int i = 0; i < 16; ++i) {
      int idx = tid + i * 128;               // dword index over 64x32
      int kr = idx >> 5, c2 = idx & 31;
      union { unsigned int u; f16 hh[2]; } cv;
      cv.u = *(const unsigned int*)(Bg + (size_t)kr * Nn + c2 * 2);
      Bt[(c2 * 2) * 64 + kr]     = cv.hh[0];
      Bt[(c2 * 2 + 1) * 64 + kr] = cv.hh[1];
    }
    // Prefetch next k-chunk rows (global_prefetch_b8)
    if (k0 + 64 < K && tid < 64) {
      __builtin_prefetch(A + (size_t)(m0 + tid) * K + k0 + 64, 0, 1);
      __builtin_prefetch(Bm + (size_t)(k0 + 64 + tid) * Nn + n0, 0, 1);
    }
    __syncthreads();
    v16h a0 = load_frag_a(As, 64, wave * 16 + l16, 0, lane);
    v16h a1 = load_frag_a(As, 64, wave * 16 + l16, 32, lane);
#pragma unroll
    for (int nf = 0; nf < 4; ++nf) {
      acc[nf] = wmma_f16(a0, load_frag_b(Bt, 64, nf * 16 + l16, 0, lane), acc[nf]);
      acc[nf] = wmma_f16(a1, load_frag_b(Bt, 64, nf * 16 + l16, 32, lane), acc[nf]);
    }
  }
  const int rowoff = (lane >= 16) ? 8 : 0;
#pragma unroll
  for (int nf = 0; nf < 4; ++nf) {
    int n = n0 + nf * 16 + l16;
#pragma unroll
    for (int r = 0; r < 8; ++r) {
      int m = m0 + wave * 16 + rowoff + r;
      if (OUT_F16) ((f16*)Cout)[(size_t)m * Nn + n]   = (f16)acc[nf][r];
      else         ((float*)Cout)[(size_t)m * Nn + n] = acc[nf][r];
    }
  }
}

// ---------------------------------------------------------------------------
// RoPE (interleaved pairs): qkv f16 [B,N,3*H*DH] -> q,k f16 [B,H,N,DH].
// q is pre-scaled by DH^-0.5 = 0.125 (exact in f16).
// ---------------------------------------------------------------------------
__global__ void k_rope(const f16* __restrict__ qkv, f16* __restrict__ qo,
                       f16* __restrict__ ko) {
  int t = blockIdx.x * blockDim.x + threadIdx.x;   // over B*H*N
  if (t >= B_ * H_ * N_) return;
  const int nseq = t % N_;
  const int bh   = t / N_;
  const int h = bh % H_, b = bh / H_;
  const f16* qrow = qkv + (size_t)(b * N_ + nseq) * (3 * H_ * DH_) + h * DH_;
  const f16* krow = qrow + H_ * DH_;
  f16* qd = qo + (size_t)(bh * N_ + nseq) * DH_;
  f16* kd = ko + (size_t)(bh * N_ + nseq) * DH_;
#pragma unroll
  for (int j = 0; j < DH_ / 2; ++j) {
    float inv = __powf(10000.0f, -(float)j / 32.0f);
    float sn, cs;
    __sincosf((float)nseq * inv, &sn, &cs);
    float q0 = (float)qrow[2 * j], q1 = (float)qrow[2 * j + 1];
    float k0 = (float)krow[2 * j], k1 = (float)krow[2 * j + 1];
    qd[2 * j]     = (f16)(0.125f * (q0 * cs - q1 * sn));
    qd[2 * j + 1] = (f16)(0.125f * (q1 * cs + q0 * sn));
    kd[2 * j]     = (f16)(k0 * cs - k1 * sn);
    kd[2 * j + 1] = (f16)(k1 * cs + k0 * sn);
  }
}

// mix[b,n,h] = sigmoid(xn . Wmix[:,h] + bmix[h]); h fastest for coalescing.
__global__ void k_mix(const f16* __restrict__ xnh, const float* __restrict__ Wmix,
                      const float* __restrict__ bmix, float* __restrict__ mix) {
  int t = blockIdx.x * blockDim.x + threadIdx.x;   // over B*N*H
  if (t >= B_ * N_ * H_) return;
  const int h = t % H_;
  const int row = t / H_;
  const f16* xr = xnh + (size_t)row * D_;
  float s = bmix[h];
  for (int d = 0; d < D_; ++d) s += (float)xr[d] * Wmix[d * H_ + h];
  mix[t] = 1.0f / (1.0f + __expf(-s));
}

// orig_v (f32, to d_out tail) + v' = v + mix*(vres - v) (f16) in [B,H,N,DH].
__global__ void k_vprep(const f16* __restrict__ qkv, const float* __restrict__ vres,
                        const float* __restrict__ mix, float* __restrict__ origv,
                        f16* __restrict__ vo) {
  int t = blockIdx.x * blockDim.x + threadIdx.x;   // over B*H*N*DH
  if (t >= B_ * H_ * N_ * DH_) return;
  const int dh = t % DH_;
  const int r  = t / DH_;         // bh*N + nseq
  const int nseq = r % N_;
  const int bh   = r / N_;
  const int h = bh % H_, b = bh / H_;
  float v = (float)qkv[(size_t)(b * N_ + nseq) * (3 * H_ * DH_) +
                       2 * H_ * DH_ + h * DH_ + dh];
  origv[t] = v;
  float m  = mix[(b * N_ + nseq) * H_ + h];
  vo[t] = (f16)(v + m * (vres[t] - v));
}

// ---------------------------------------------------------------------------
// Block-causal flash attention. Block = (b,h) x 64 queries (4 waves x 16 rows);
// query block of 64 lies inside one 256-causal-block => uniform key range
// kmax = (q0/256+1)*256, no element masking. Key tiles of 64 staged in LDS
// (16 WMMA per stage round). launch_bounds(128,1): max VGPR budget so Q frags
// and O accumulators stay register-resident (no scratch spills).
// ---------------------------------------------------------------------------
__global__ __launch_bounds__(128, 1)
void k_attn(const f16* __restrict__ q, const f16* __restrict__ k,
            const f16* __restrict__ v, f16* __restrict__ ao) {
  __shared__ f16 Ks[64 * 64];        // [key][dh]
  __shared__ f16 Vt[64 * 64];        // [dh][key]  (transposed for P.V)
  __shared__ f16 Ps[4][16 * 64];     // per-wave P tile [row][key]
  const int tid  = threadIdx.x;      // 128
  const int lane = tid & 31;
  const int wave = tid >> 5;
  const int l16  = lane & 15;
  const int bh = blockIdx.x;
  const int b = bh / H_, h = bh % H_;
  const int q0   = blockIdx.y * 64;
  const int qw   = q0 + wave * 16;
  const int kmax = ((q0 / BS_) + 1) * BS_;
  const f16* qbase = q + (size_t)bh * N_ * DH_;
  const f16* kbase = k + (size_t)bh * N_ * DH_;
  const f16* vbase = v + (size_t)bh * N_ * DH_;

  v16h qf0 = load_frag_a(qbase, DH_, qw + l16, 0, lane);
  v16h qf1 = load_frag_a(qbase, DH_, qw + l16, 32, lane);
  v8f oacc[4] = {};
  float mrow[8], lrow[8];
#pragma unroll
  for (int r = 0; r < 8; ++r) { mrow[r] = -1e30f; lrow[r] = 0.f; }

  for (int kt = 0; kt < kmax; kt += 64) {
    __syncthreads();
    // Stage K tile [64 keys][64 dh] (2048 dwords, coalesced)
    const unsigned int* Kg = (const unsigned int*)(kbase + (size_t)kt * DH_);
#pragma unroll
    for (int i = 0; i < 16; ++i) {
      int idx = tid + i * 128;
      ((unsigned int*)Ks)[idx] = Kg[idx];
    }
    // Stage V transposed: read [key][dh] coalesced, write [dh][key]
    const f16* Vg = vbase + (size_t)kt * DH_;
#pragma unroll
    for (int i = 0; i < 16; ++i) {
      int idx = tid + i * 128;
      int kr = idx >> 5, c2 = idx & 31;
      union { unsigned int u; f16 hh[2]; } cv;
      cv.u = *(const unsigned int*)(Vg + (size_t)kr * DH_ + c2 * 2);
      Vt[(c2 * 2) * 64 + kr]     = cv.hh[0];
      Vt[(c2 * 2 + 1) * 64 + kr] = cv.hh[1];
    }
    // Prefetch next K/V tile (global_prefetch_b8): 128 threads x 64B = 8KB each
    if (kt + 64 < kmax) {
      __builtin_prefetch((const char*)(kbase + (size_t)(kt + 64) * DH_) + tid * 64, 0, 1);
      __builtin_prefetch((const char*)(vbase + (size_t)(kt + 64) * DH_) + tid * 64, 0, 1);
    }
    __syncthreads();
    // S = (q*scale) . K^T  for 64 keys (four 16-key column groups)
    v8f s[4];
#pragma unroll
    for (int nf = 0; nf < 4; ++nf) {
      v8f t = {};
      t = wmma_f16(qf0, load_frag_b(Ks, 64, nf * 16 + l16, 0, lane), t);
      t = wmma_f16(qf1, load_frag_b(Ks, 64, nf * 16 + l16, 32, lane), t);
      s[nf] = t;
    }
    // Online softmax (row = r + 8*(lane>=16); 16-lane row reductions)
#pragma unroll
    for (int r = 0; r < 8; ++r) {
      float pm = fmaxf(fmaxf(s[0][r], s[1][r]), fmaxf(s[2][r], s[3][r]));
#pragma unroll
      for (int msk = 8; msk; msk >>= 1) pm = fmaxf(pm, __shfl_xor(pm, msk, 32));
      float mnew = fmaxf(mrow[r], pm);
      float corr = __expf(mrow[r] - mnew);
      mrow[r] = mnew;
      float rs = 0.f;
#pragma unroll
      for (int nf = 0; nf < 4; ++nf) {
        float p = __expf(s[nf][r] - mnew);
        s[nf][r] = p;
        rs += p;
      }
#pragma unroll
      for (int msk = 8; msk; msk >>= 1) rs += __shfl_xor(rs, msk, 32);
      lrow[r] = lrow[r] * corr + rs;
#pragma unroll
      for (int j = 0; j < 4; ++j) oacc[j][r] *= corr;
    }
    // C-layout -> A-layout for P via per-wave LDS round trip
    f16* P = &Ps[wave][0];
    const int rowoff = (lane >= 16) ? 8 : 0;
#pragma unroll
    for (int r = 0; r < 8; ++r) {
#pragma unroll
      for (int nf = 0; nf < 4; ++nf)
        P[(rowoff + r) * 64 + nf * 16 + l16] = (f16)s[nf][r];
    }
    v16h pf0 = load_frag_a(P, 64, l16, 0, lane);
    v16h pf1 = load_frag_a(P, 64, l16, 32, lane);
#pragma unroll
    for (int j = 0; j < 4; ++j) {
      oacc[j] = wmma_f16(pf0, load_frag_b(Vt, 64, j * 16 + l16, 0, lane), oacc[j]);
      oacc[j] = wmma_f16(pf1, load_frag_b(Vt, 64, j * 16 + l16, 32, lane), oacc[j]);
    }
  }
  // Epilogue: normalize and write f16 attn-out [B,N,H*DH]
  const int rowoff = (lane >= 16) ? 8 : 0;
#pragma unroll
  for (int j = 0; j < 4; ++j) {
    int dh = j * 16 + l16;
#pragma unroll
    for (int r = 0; r < 8; ++r) {
      int qrow = qw + rowoff + r;
      float val = oacc[j][r] / lrow[r];
      ao[(size_t)(b * N_ + qrow) * (H_ * DH_) + h * DH_ + dh] = (f16)val;
    }
  }
}

// ---------------------------------------------------------------------------
extern "C" void kernel_launch(void* const* d_in, const int* in_sizes, int n_in,
                              void* d_out, int out_size, void* d_ws, size_t ws_size,
                              hipStream_t stream) {
  const float* x     = (const float*)d_in[0];
  const float* vres  = (const float*)d_in[1];
  const float* gamma = (const float*)d_in[2];
  const float* Wqkv  = (const float*)d_in[3];
  const float* Wout  = (const float*)d_in[4];
  const float* Wmix  = (const float*)d_in[5];
  const float* bmix  = (const float*)d_in[6];

  float* out   = (float*)d_out;                       // [B,N,D]
  float* origv = out + (size_t)B_ * N_ * D_;          // [B,H,N,DH]

  const size_t nTok = (size_t)B_ * N_;                // 4096
  const size_t nHead = (size_t)B_ * H_ * N_;          // 65536
  char* ws = (char*)d_ws;
  f16* xnh    = (f16*)ws;                 ws += nTok * D_ * 2;                 // 8MB
  f16* wqkvh  = (f16*)ws;                 ws += (size_t)D_ * 3 * H_ * DH_ * 2; // 6MB
  f16* wouth  = (f16*)ws;                 ws += (size_t)H_ * DH_ * D_ * 2;     // 2MB
  f16* qkvh   = (f16*)ws;                 ws += nTok * 3 * H_ * DH_ * 2;       // 24MB
  f16* qh     = (f16*)ws;                 ws += nHead * DH_ * 2;               // 8MB
  f16* kh     = (f16*)ws;                 ws += nHead * DH_ * 2;               // 8MB
  f16* vh     = (f16*)ws;                 ws += nHead * DH_ * 2;               // 8MB
  float* mixf = (float*)ws;               ws += nTok * H_ * 4;                 // 256KB
  f16* aoh    = (f16*)ws;                 ws += nTok * H_ * DH_ * 2;           // 8MB

  // 1) RMSNorm -> f16
  k_rmsnorm<<<B_ * N_, 256, 0, stream>>>(x, gamma, xnh);
  // 2) weights -> f16
  {
    int n1 = D_ * 3 * H_ * DH_;
    k_f32_to_f16<<<(n1 + 255) / 256, 256, 0, stream>>>(Wqkv, wqkvh, n1);
    int n2 = H_ * DH_ * D_;
    k_f32_to_f16<<<(n2 + 255) / 256, 256, 0, stream>>>(Wout, wouth, n2);
  }
  // 3) QKV GEMM (WMMA): [4096,1024] x [1024,3072] -> f16
  k_gemm<true><<<dim3((3 * H_ * DH_) / 64, (int)nTok / 64), 128, 0, stream>>>(
      xnh, wqkvh, qkvh, (int)nTok, D_, 3 * H_ * DH_);
  // 4) RoPE on q,k (q pre-scaled)
  k_rope<<<((int)nHead + 255) / 256, 256, 0, stream>>>(qkvh, qh, kh);
  // 5) mix = sigmoid(xn @ Wmix + bmix)
  k_mix<<<((int)(nTok * H_) + 127) / 128, 128, 0, stream>>>(xnh, Wmix, bmix, mixf);
  // 6) orig_v out + v' = lerp(v, vres, mix)
  k_vprep<<<((int)(nHead * DH_) + 255) / 256, 256, 0, stream>>>(qkvh, vres, mixf,
                                                                origv, vh);
  // 7) block-causal flash attention (WMMA)
  k_attn<<<dim3(B_ * H_, N_ / 64), 128, 0, stream>>>(qh, kh, vh, aoh);
  // 8) out projection (WMMA): [4096,1024] x [1024,1024] -> f32 d_out
  k_gemm<false><<<dim3(D_ / 64, (int)nTok / 64), 128, 0, stream>>>(
      aoh, wouth, out, (int)nTok, H_ * DH_, D_);
  (void)in_sizes; (void)n_in; (void)out_size; (void)ws_size;
}
